// SelfAttention_33406255628855
// MI455X (gfx1250) — matrix operand order repaired and checked by reading, not compiled, not run on hardware
//
#include <hip/hip_runtime.h>

// Self-attention, MI455X (gfx1250), wave32 + WMMA bf16 + Tensor Data Mover.
// Pipeline: [convert x] [convert+transpose W] -> [qkv gemm (TDM)] ->
//           [flash attention] -> [out gemm (TDM)]

#define B_   32
#define S_   500
#define C_   1024
#define H_   16
#define D_   64
#define SP_  512            // S padded to multiple of 128
#define MTOK (B_ * S_)      // 16000 tokens (divisible by 128)

typedef __attribute__((ext_vector_type(16))) __bf16 bf16x16;
typedef __attribute__((ext_vector_type(8)))  __bf16 bf16x8;
typedef __attribute__((ext_vector_type(8)))  float  f32x8;
typedef __attribute__((ext_vector_type(4)))  unsigned int u32x4;
typedef __attribute__((ext_vector_type(8)))  int i32x8;
typedef __attribute__((ext_vector_type(4)))  int i32x4;

static __device__ __forceinline__ bf16x8 ld8(const __bf16* p) {
  return *(const bf16x8*)p;
}
static __device__ __forceinline__ bf16x16 mk16(bf16x8 lo, bf16x8 hi) {
  union { bf16x16 v; bf16x8 h[2]; } u;
  u.h[0] = lo; u.h[1] = hi;
  return u.v;
}
static __device__ __forceinline__ bf16x16 ld16(const __bf16* p) {
  return mk16(ld8(p), ld8(p + 8));
}
static __device__ __forceinline__ f32x8 zero8() {
  f32x8 z = {0.f, 0.f, 0.f, 0.f, 0.f, 0.f, 0.f, 0.f};
  return z;
}
static __device__ __forceinline__ f32x8 wmma_bf16(bf16x16 a, bf16x16 b, f32x8 c) {
  return __builtin_amdgcn_wmma_f32_16x16x32_bf16(false, a, false, b, (short)0, c,
                                                 false, false);
}
static __device__ __forceinline__ float hmax16(float v) {
  v = fmaxf(v, __shfl_xor(v, 1, 32));
  v = fmaxf(v, __shfl_xor(v, 2, 32));
  v = fmaxf(v, __shfl_xor(v, 4, 32));
  v = fmaxf(v, __shfl_xor(v, 8, 32));
  return v;
}
static __device__ __forceinline__ float hsum16(float v) {
  v += __shfl_xor(v, 1, 32);
  v += __shfl_xor(v, 2, 32);
  v += __shfl_xor(v, 4, 32);
  v += __shfl_xor(v, 8, 32);
  return v;
}

// ---------------------------------------------------------------------------
// TDM: DMA a 2-D bf16 tile [tile_d1 rows x tile_d0 cols] from global memory
// (row stride = stride elems) into LDS at lds_addr, inserting 16 bytes of
// padding after every 64-byte row so LDS row stride = 40 bf16 (matches the
// WMMA fragment loaders).  D# bit layout per CDNA5 ISA ch.10.8.
// ---------------------------------------------------------------------------
static __device__ __forceinline__ void tdm_load_2d(unsigned lds_addr,
                                                   const void* gaddr,
                                                   unsigned tile_d0,
                                                   unsigned tile_d1,
                                                   unsigned td0, unsigned td1,
                                                   unsigned stride) {
  const unsigned long long ga = (unsigned long long)(size_t)gaddr;
  u32x4 g0;
  g0[0] = 1u;                                   // count=1, user mode
  g0[1] = lds_addr;                             // LDS byte address
  g0[2] = (unsigned)(ga & 0xffffffffu);         // global_addr[31:0]
  g0[3] = (unsigned)((ga >> 32) & 0x01ffffffu)  // global_addr[56:32]
          | (2u << 30);                         // type=2 ("image")
  i32x8 g1;
  g1[0] = (int)((1u << 16)      // data_size = 1 (2 bytes)
                | (1u << 20)    // pad_enable
                | (3u << 22)    // pad_interval: every 16 DWORDs (64B)
                | (3u << 25));  // pad_amount: 4 DWORDs (16B)
  g1[1] = (int)((td0 & 0xffffu) << 16);                          // dim0 lo
  g1[2] = (int)(((td0 >> 16) & 0xffffu) | ((td1 & 0xffffu) << 16));
  g1[3] = (int)(((td1 >> 16) & 0xffffu) | (tile_d0 << 16));
  g1[4] = (int)(tile_d1 & 0xffffu);  // tile_dim1 (tile_dim2 = 0)
  g1[5] = (int)stride;               // tensor_dim0_stride[31:0]
  g1[6] = 0;
  g1[7] = 0;
  const i32x4 z4 = {0, 0, 0, 0};
#if defined(__clang_major__) && __clang_major__ >= 23
  const i32x8 z8 = {0, 0, 0, 0, 0, 0, 0, 0};
  __builtin_amdgcn_tensor_load_to_lds(g0, g1, z4, z4, z8, 0);
#else
  __builtin_amdgcn_tensor_load_to_lds(g0, g1, z4, z4, 0);
#endif
}

// ---------------------------------------------------------------------------
// Pre-pass: x fp32 -> bf16 (vectorized 8/thread)
// ---------------------------------------------------------------------------
__global__ __launch_bounds__(256)
void convert_x_kernel(const float* __restrict__ x, __bf16* __restrict__ Xb) {
  const size_t i = ((size_t)blockIdx.x * 256 + threadIdx.x) * 8;
  const float4 a = *(const float4*)(x + i);
  const float4 b = *(const float4*)(x + i + 4);
  bf16x8 o;
  o[0] = (__bf16)a.x; o[1] = (__bf16)a.y; o[2] = (__bf16)a.z; o[3] = (__bf16)a.w;
  o[4] = (__bf16)b.x; o[5] = (__bf16)b.y; o[6] = (__bf16)b.z; o[7] = (__bf16)b.w;
  *(bf16x8*)(Xb + i) = o;
}

// ---------------------------------------------------------------------------
// Pre-pass: W fp32 [k][n] -> bf16 transposed T[n][k], 32x32 LDS tiles.
// blockIdx.z selects which of the 4 weight matrices.
// ---------------------------------------------------------------------------
__global__ __launch_bounds__(256)
void transpose_w_kernel(const float* __restrict__ W0, const float* __restrict__ W1,
                        const float* __restrict__ W2, const float* __restrict__ W3,
                        __bf16* __restrict__ T0, __bf16* __restrict__ T1,
                        __bf16* __restrict__ T2, __bf16* __restrict__ T3) {
  const int z = blockIdx.z;
  const float* W = (z == 0) ? W0 : ((z == 1) ? W1 : ((z == 2) ? W2 : W3));
  __bf16* T = (z == 0) ? T0 : ((z == 1) ? T1 : ((z == 2) ? T2 : T3));
  __shared__ float tile[32][33];
  const int t = threadIdx.x;
  const int col = t & 31;
  const int row4 = t >> 5;  // 0..7
  const int n0 = blockIdx.x * 32;
  const int k0 = blockIdx.y * 32;
#pragma unroll
  for (int i = 0; i < 4; ++i) {
    const int r = row4 * 4 + i;
    tile[r][col] = W[(size_t)(k0 + r) * C_ + n0 + col];
  }
  __syncthreads();
#pragma unroll
  for (int i = 0; i < 4; ++i) {
    const int r = row4 * 4 + i;
    T[(size_t)(n0 + r) * C_ + k0 + col] = (__bf16)tile[col][r];
  }
}

// ---------------------------------------------------------------------------
// GEMM via TDM double-buffering.  out = A(bf16) @ Bt^T + bias.
// A: [Mtok x 1024] bf16 row-major.  Bt: [n][k] bf16 (pre-transposed).
// Block tile 128(M) x 64(N), K-step 32; wave 0 drives the TDM, 8 waves
// each compute a 32x32 tile (4 WMMA accumulators).
// mode 0/1/2: scatter bf16 into Q/K [B,H,Spad,D] or V^T [B,H,D,Spad].
// mode 3:     fp32 store to out.
// ---------------------------------------------------------------------------
__global__ __launch_bounds__(256)
void gemm_tdm_kernel(const __bf16* __restrict__ Ag,
                     const __bf16* __restrict__ Bt0, const float* __restrict__ bias0,
                     const __bf16* __restrict__ Bt1, const float* __restrict__ bias1,
                     const __bf16* __restrict__ Bt2, const float* __restrict__ bias2,
                     int mode_sel,
                     __bf16* __restrict__ Qb, __bf16* __restrict__ Kb,
                     __bf16* __restrict__ Vt, float* __restrict__ out) {
  const int t    = threadIdx.x;
  const int lane = t & 31;
  const int wave = t >> 5;
  const int lmod = lane & 15;
  const int lhi  = lane >> 4;
  const int wm   = wave & 3;   // M offset wm*32
  const int wn   = wave >> 2;  // N offset wn*32
  const int mode = (mode_sel >= 0) ? mode_sel : (int)blockIdx.z;
  const __bf16* Btg  = (mode == 1) ? Bt1 : ((mode == 2) ? Bt2 : Bt0);
  const float*  bias = (mode == 1) ? bias1 : ((mode == 2) ? bias2 : bias0);
  const int m0 = blockIdx.y * 128;
  const int n0 = blockIdx.x * 64;

  __shared__ __align__(16) __bf16 As[2][128][40];  // stride 40 = TDM pad
  __shared__ __align__(16) __bf16 Bs[2][64][40];

  f32x8 acc[2][2];
#pragma unroll
  for (int mi = 0; mi < 2; ++mi)
#pragma unroll
    for (int ni = 0; ni < 2; ++ni) acc[mi][ni] = zero8();

  const __bf16* Abase = Ag + (size_t)m0 * C_;
  const __bf16* Bbase = Btg + (size_t)n0 * C_;
  const unsigned ldsA[2] = {(unsigned)(size_t)&As[0][0][0],
                            (unsigned)(size_t)&As[1][0][0]};
  const unsigned ldsB[2] = {(unsigned)(size_t)&Bs[0][0][0],
                            (unsigned)(size_t)&Bs[1][0][0]};

  if (wave == 0) {  // prologue: fill buffer 0
    tdm_load_2d(ldsA[0], Abase, 32, 128, C_, MTOK, C_);
    tdm_load_2d(ldsB[0], Bbase, 32, 64, C_, C_, C_);
  }

  for (int it = 0; it < C_ / 32; ++it) {
    const int cur = it & 1;
    if (wave == 0) __builtin_amdgcn_s_wait_tensorcnt(0);
    __syncthreads();  // buffer[cur] ready for everyone
    if (wave == 0 && it + 1 < C_ / 32) {  // async-prefetch next K-slice
      const int k0n = (it + 1) * 32;
      tdm_load_2d(ldsA[cur ^ 1], Abase + k0n, 32, 128, C_, MTOK, C_);
      tdm_load_2d(ldsB[cur ^ 1], Bbase + k0n, 32, 64, C_, C_, C_);
    }

    bf16x16 af[2], bfr[2];
#pragma unroll
    for (int mi = 0; mi < 2; ++mi) {
      const __bf16* p = &As[cur][wm * 32 + mi * 16 + lmod][lhi * 16];
      af[mi] = mk16(ld8(p), ld8(p + 8));
    }
#pragma unroll
    for (int ni = 0; ni < 2; ++ni)
      bfr[ni] = ld16(&Bs[cur][wn * 32 + ni * 16 + lmod][lhi * 16]);
#pragma unroll
    for (int mi = 0; mi < 2; ++mi)
#pragma unroll
      for (int ni = 0; ni < 2; ++ni)
        acc[mi][ni] = wmma_bf16(af[mi], bfr[ni], acc[mi][ni]);

    __syncthreads();  // everyone done with buffer[cur] before it is re-filled
  }

  // ---- epilogue ----
#pragma unroll
  for (int mi = 0; mi < 2; ++mi)
#pragma unroll
    for (int ni = 0; ni < 2; ++ni)
#pragma unroll
      for (int r = 0; r < 8; ++r) {
        const int rowg = m0 + wm * 32 + mi * 16 + r + 8 * lhi;  // token
        const int colg = n0 + wn * 32 + ni * 16 + lmod;         // channel
        const float v  = acc[mi][ni][r] + bias[colg];
        if (mode == 3) {
          out[(size_t)rowg * C_ + colg] = v;
        } else {
          const int bb = rowg / S_;
          const int ss = rowg - bb * S_;
          const int hh = colg >> 6;
          const int dd = colg & 63;
          if (mode == 0)
            Qb[((size_t)(bb * H_ + hh) * SP_ + ss) * D_ + dd] = (__bf16)v;
          else if (mode == 1)
            Kb[((size_t)(bb * H_ + hh) * SP_ + ss) * D_ + dd] = (__bf16)v;
          else
            Vt[((size_t)(bb * H_ + hh) * D_ + dd) * SP_ + ss] = (__bf16)v;
        }
      }
}

// ---------------------------------------------------------------------------
// Flash attention: one wave owns 16 query rows x full D=64 output.
// Online softmax; per-row stats replicated across each 16-lane half.
// P tile is re-shaped C/D->A layout via a per-wave LDS round trip.
// Causal mask also hides the S->SP_ padding of K/V.
// ---------------------------------------------------------------------------
__global__ __launch_bounds__(256)
void attn_kernel(const __bf16* __restrict__ Qb, const __bf16* __restrict__ Kb,
                 const __bf16* __restrict__ Vt, __bf16* __restrict__ Y) {
  const int lane  = threadIdx.x & 31;
  const int wave  = threadIdx.x >> 5;
  const int lmod  = lane & 15;
  const int lhi   = lane >> 4;
  const int h     = blockIdx.y;
  const int b     = blockIdx.z;
  const int bh    = b * H_ + h;
  const int qbase = blockIdx.x * 128 + wave * 16;

  __shared__ __align__(16) __bf16 Ps[8][16][40];  // per-wave P tile

  const __bf16* qrow = Qb + ((size_t)bh * SP_ + qbase + lmod) * D_;
  bf16x16 aq[2];
#pragma unroll
  for (int d32 = 0; d32 < 2; ++d32) {
    const __bf16* p = qrow + d32 * 32 + lhi * 16;
    aq[d32] = mk16(ld8(p), ld8(p + 8));
  }

  f32x8 o[4];
#pragma unroll
  for (int i = 0; i < 4; ++i) o[i] = zero8();
  float mst[8], lst[8];
#pragma unroll
  for (int r = 0; r < 8; ++r) { mst[r] = -1e30f; lst[r] = 0.f; }

  const int nsteps = (qbase + 16 + 31) >> 5;  // causal: only needed key tiles
  for (int j = 0; j < nsteps; ++j) {
    const int jb = j * 32;

    // ---- scores: S = Q @ K^T ----
    f32x8 sc[2];
#pragma unroll
    for (int h2 = 0; h2 < 2; ++h2) {
      sc[h2] = zero8();
      const int key = jb + h2 * 16 + lmod;
      const __bf16* krow = Kb + ((size_t)bh * SP_ + key) * D_;
#pragma unroll
      for (int d32 = 0; d32 < 2; ++d32) {
        bf16x16 bk = ld16(krow + d32 * 32 + lhi * 16);
        sc[h2] = wmma_bf16(aq[d32], bk, sc[h2]);
      }
    }

    // ---- online softmax update ----
#pragma unroll
    for (int r = 0; r < 8; ++r) {
      const int row = qbase + r + 8 * lhi;
      float s0 = sc[0][r] * 0.125f;  // 1/sqrt(64)
      float s1 = sc[1][r] * 0.125f;
      if (jb + lmod      > row) s0 = -1e30f;
      if (jb + 16 + lmod > row) s1 = -1e30f;
      const float tmax = hmax16(fmaxf(s0, s1));
      const float mnew = fmaxf(mst[r], tmax);
      const float corr = __expf(mst[r] - mnew);
      mst[r] = mnew;
      const float p0 = __expf(s0 - mnew);
      const float p1 = __expf(s1 - mnew);
      lst[r] = lst[r] * corr + hsum16(p0 + p1);
#pragma unroll
      for (int d16 = 0; d16 < 4; ++d16) o[d16][r] *= corr;
      Ps[wave][r + 8 * lhi][lmod]      = (__bf16)p0;
      Ps[wave][r + 8 * lhi][16 + lmod] = (__bf16)p1;
    }
    asm volatile("s_wait_dscnt 0" ::: "memory");  // same-wave LDS RAW fence

    // ---- O += P @ V ----
    bf16x16 ap;
    {
      const __bf16* p = &Ps[wave][lmod][lhi * 16];
      ap = mk16(ld8(p), ld8(p + 8));
    }
#pragma unroll
    for (int d16 = 0; d16 < 4; ++d16) {
      const __bf16* vrow =
          Vt + ((size_t)bh * D_ + d16 * 16 + lmod) * SP_ + jb + lhi * 16;
      o[d16] = wmma_bf16(ap, ld16(vrow), o[d16]);
    }
  }

  // ---- epilogue: normalize, store to Y [token, C] bf16 ----
#pragma unroll
  for (int r = 0; r < 8; ++r) {
    const int ss = qbase + r + 8 * lhi;
    if (ss >= S_) continue;
    const float rinv = 1.0f / lst[r];
#pragma unroll
    for (int d16 = 0; d16 < 4; ++d16)
      Y[((size_t)(b * S_ + ss)) * C_ + h * D_ + d16 * 16 + lmod] =
          (__bf16)(o[d16][r] * rinv);
  }
}

// ---------------------------------------------------------------------------
extern "C" void kernel_launch(void* const* d_in, const int* in_sizes, int n_in,
                              void* d_out, int out_size, void* d_ws,
                              size_t ws_size, hipStream_t stream) {
  const float* x  = (const float*)d_in[0];
  const float* Wq = (const float*)d_in[1];
  const float* bq = (const float*)d_in[2];
  const float* Wk = (const float*)d_in[3];
  const float* bk = (const float*)d_in[4];
  const float* Wv = (const float*)d_in[5];
  const float* bv = (const float*)d_in[6];
  const float* Wp = (const float*)d_in[7];
  const float* bp = (const float*)d_in[8];
  float* out = (float*)d_out;

  // Workspace (bf16): Xb/Yb (aliased, 32MB) | Q | K | V^T | 4x W^T  ~= 142 MB
  const size_t nx  = (size_t)MTOK * C_;          // 16,384,000
  const size_t nqk = (size_t)B_ * H_ * SP_ * D_; // 16,777,216
  const size_t nw  = (size_t)C_ * C_;            // 1,048,576
  __bf16* Xb  = (__bf16*)d_ws;  // doubles as Yb after the QKV GEMM
  __bf16* Qb  = Xb + nx;
  __bf16* Kb  = Qb + nqk;
  __bf16* Vt  = Kb + nqk;
  __bf16* Wtq = Vt + nqk;
  __bf16* Wtk = Wtq + nw;
  __bf16* Wtv = Wtk + nw;
  __bf16* Wtp = Wtv + nw;

  dim3 blk(256);
  convert_x_kernel<<<dim3((unsigned)(nx / (256 * 8))), blk, 0, stream>>>(x, Xb);
  transpose_w_kernel<<<dim3(32, 32, 4), blk, 0, stream>>>(Wq, Wk, Wv, Wp, Wtq,
                                                          Wtk, Wtv, Wtp);
  gemm_tdm_kernel<<<dim3(C_ / 64, MTOK / 128, 3), blk, 0, stream>>>(
      Xb, Wtq, bq, Wtk, bk, Wtv, bv, /*mode_sel=*/-1, Qb, Kb, Vt, nullptr);
  attn_kernel<<<dim3(SP_ / 128, H_, B_), blk, 0, stream>>>(Qb, Kb, Vt, Xb);
  gemm_tdm_kernel<<<dim3(C_ / 64, MTOK / 128, 1), blk, 0, stream>>>(
      Xb, Wtp, bp, Wtp, bp, Wtp, bp, /*mode_sel=*/3, nullptr, nullptr, nullptr,
      out);
}
